// KMeansVectorQuantizer_52123723105002
// MI455X (gfx1250) — matrix operand order repaired and checked by reading, not compiled, not run on hardware
//
#include <hip/hip_runtime.h>
#include <hip/hip_bf16.h>

typedef __attribute__((ext_vector_type(16))) __bf16 v16bf;
typedef __attribute__((ext_vector_type(8)))  float  v8f;

#define NUM_EMBED 1024
#define EMBED_DIM 256
#define TDIM 4096
#define BDIM 16
#define NROWS (BDIM * TDIM)            // 65536 flattened vectors
#define ROWS_PER_BLOCK 64
#define THREADS 128
#define LDK 264                        // padded bf16 row pitch (528 B) -> conflict-free b128 reads
#define OUT_MAIN (BDIM * EMBED_DIM * TDIM)   // 16777216 floats of z_q output

// ---------------------------------------------------------------------------
// Prep: codebook squared norms (exact fp32), zero histogram + loss partials
// ---------------------------------------------------------------------------
__global__ __launch_bounds__(256) void vq_prep(const float* __restrict__ embed,
                                               float* __restrict__ enorm,
                                               float* __restrict__ partials,
                                               int* __restrict__ hist) {
    int c = blockIdx.x * blockDim.x + threadIdx.x;
    if (c < NUM_EMBED) {
        const float* row = embed + (size_t)c * EMBED_DIM;
        float s = 0.f;
        for (int d = 0; d < EMBED_DIM; ++d) { float v = row[d]; s += v * v; }
        enorm[c] = s;
        hist[c] = 0;
        partials[c] = 0.f;             // exactly NROWS/ROWS_PER_BLOCK == 1024 partials
    }
}

// ---------------------------------------------------------------------------
// Main: fused distance GEMM on the bf16 matrix pipe (split hi/lo for fp32-class
// accuracy) + register argmin + gather + loss.
// Block = 128 threads = 4 waves; each wave owns 16 rows; block owns 64 rows.
// ---------------------------------------------------------------------------
__global__ __launch_bounds__(THREADS) void vq_main(const float* __restrict__ inp,
                                                   const float* __restrict__ embed,
                                                   const float* __restrict__ enorm,
                                                   float* __restrict__ out,
                                                   float* __restrict__ partials,
                                                   int* __restrict__ hist) {
    __shared__ __bf16 AsH[ROWS_PER_BLOCK * LDK];   // 33 KB  x_hi tile (64 x 256)
    __shared__ __bf16 AsL[ROWS_PER_BLOCK * LDK];   // 33 KB  x_lo tile
    __shared__ __bf16 BsH[16 * LDK];               // 8.3 KB e_hi tile (16 codes)
    __shared__ __bf16 BsL[16 * LDK];               // 8.3 KB e_lo tile
    __shared__ float  xnp[THREADS];
    __shared__ float  xnorm_s[ROWS_PER_BLOCK];
    __shared__ int    qidx_s[ROWS_PER_BLOCK];
    __shared__ float  red[THREADS];

    const int tid      = threadIdx.x;
    const int lane     = tid & 31;
    const int waveBase = (tid >> 5) << 4;        // wave w -> local rows [16w, 16w+16)
    const int n0 = blockIdx.x * ROWS_PER_BLOCK;  // flattened row base (b*T + t)
    const int b  = n0 >> 12;                     // / 4096
    const int t0 = n0 & (TDIM - 1);
    const size_t bBase = (size_t)b * EMBED_DIM * TDIM;

    // ---- Stage A tile: inputs is (B, D, T); split fp32 -> bf16 hi + bf16 lo.
    // Exact fp32 row norms accumulated on the fly.
    {
        const int tl = tid & 63;
        const int dp = tid >> 6;                 // 0 or 1
        float xa = 0.f;
        #pragma unroll 4
        for (int it = 0; it < 128; ++it) {
            int d = it * 2 + dp;
            float v = inp[bBase + (size_t)d * TDIM + t0 + tl];
            __bf16 h = (__bf16)v;
            AsH[tl * LDK + d] = h;
            AsL[tl * LDK + d] = (__bf16)(v - (float)h);
            xa += v * v;
        }
        xnp[tid] = xa;
    }
    __syncthreads();
    if (tid < ROWS_PER_BLOCK) xnorm_s[tid] = xnp[tid] + xnp[tid + 64];
    __syncthreads();

    // Per-lane fragment bases per ISA 16-bit layouts:
    //  A 16x32: lanes 0-15 -> K runs {0..7, 16..23}; lanes 16-31 -> {8..15, 24..31}
    //  B 32x16: lanes 0-15 -> K runs {0..7, 16..23}; lanes 16-31 -> {8..15, 24..31}
    const int m    = lane & 15;
    const int half = lane >> 4;
    const __bf16* aHp = &AsH[(waveBase + m) * LDK + half * 8];
    const __bf16* aLp = &AsL[(waveBase + m) * LDK + half * 8];
    const __bf16* bHp = &BsH[m * LDK + half * 8];
    const __bf16* bLp = &BsL[m * LDK + half * 8];

    float xn[8];
    #pragma unroll
    for (int j = 0; j < 8; ++j) xn[j] = xnorm_s[waveBase + half * 8 + j];

    float bestD[8];
    int   bestI[8];
    #pragma unroll
    for (int j = 0; j < 8; ++j) { bestD[j] = 3.4e38f; bestI[j] = 0; }

    for (int ct = 0; ct < NUM_EMBED / 16; ++ct) {
        __syncthreads();
        // Stage 16 codebook rows, split hi/lo (coalesced, shared by all waves).
        #pragma unroll
        for (int itb = 0; itb < 32; ++itb) {
            int e  = itb * THREADS + tid;
            int cl = e >> 8;
            int d  = e & 255;
            float v = embed[(size_t)(ct * 16 + cl) * EMBED_DIM + d];
            __bf16 h = (__bf16)v;
            BsH[cl * LDK + d] = h;
            BsL[cl * LDK + d] = (__bf16)(v - (float)h);
        }
        if (ct < NUM_EMBED / 16 - 1 && tid == 0)
            __builtin_prefetch(&embed[(size_t)((ct + 1) * 16) * EMBED_DIM], 0, 1);
        __syncthreads();

        // dot(x, e) ~= xh.eh + xh.el + xl.eh  (error ~2^-16, f32 accumulate)
        v8f acc = {};
        #pragma unroll 4
        for (int kc = 0; kc < EMBED_DIM / 32; ++kc) {
            v16bf ah, al, bh, bl;
            const __bf16* pa = aHp + kc * 32;
            __builtin_memcpy(&ah, pa, 16);
            __builtin_memcpy((char*)&ah + 16, pa + 16, 16);
            const __bf16* pl = aLp + kc * 32;
            __builtin_memcpy(&al, pl, 16);
            __builtin_memcpy((char*)&al + 16, pl + 16, 16);
            __builtin_memcpy(&bh, bHp + kc * 32, 16);
            __builtin_memcpy((char*)&bh + 16, bHp + kc * 32 + 16, 16);
            __builtin_memcpy(&bl, bLp + kc * 32, 16);
            __builtin_memcpy((char*)&bl + 16, bLp + kc * 32 + 16, 16);

            acc = __builtin_amdgcn_wmma_f32_16x16x32_bf16(false, ah, false, bh,
                                                          (short)0, acc, false, false);
            acc = __builtin_amdgcn_wmma_f32_16x16x32_bf16(false, ah, false, bl,
                                                          (short)0, acc, false, false);
            acc = __builtin_amdgcn_wmma_f32_16x16x32_bf16(false, al, false, bh,
                                                          (short)0, acc, false, false);
        }

        const float en  = enorm[ct * 16 + m];
        const int   idx = ct * 16 + m;
        #pragma unroll
        for (int j = 0; j < 8; ++j) {
            float d2 = xn[j] + en - 2.0f * acc[j];     // ||x||^2 + ||e||^2 - 2 x.e
            if (d2 < bestD[j]) { bestD[j] = d2; bestI[j] = idx; }
        }
    }

    // Cross-lane argmin: butterfly over the 16 lanes of each C half (wave32).
    #pragma unroll
    for (int mk = 1; mk < 16; mk <<= 1) {
        #pragma unroll
        for (int j = 0; j < 8; ++j) {
            float od = __shfl_xor(bestD[j], mk, 32);
            int   oi = __shfl_xor(bestI[j], mk, 32);
            if (od < bestD[j] || (od == bestD[j] && oi < bestI[j])) {
                bestD[j] = od; bestI[j] = oi;
            }
        }
    }
    if (m == 0) {
        #pragma unroll
        for (int j = 0; j < 8; ++j)
            qidx_s[waveBase + half * 8 + j] = bestI[j];
    }
    __syncthreads();

    // Codebook-usage histogram (int atomics: deterministic).
    if (tid < ROWS_PER_BLOCK) atomicAdd(&hist[qidx_s[tid]], 1);

    // Gather winning code rows (L2-resident), coalesced stores to (B,D,T);
    // loss partial against the exact fp32 inputs (streamed re-read).
    {
        const int tl = tid & 63;
        const int dp = tid >> 6;
        const int q  = qidx_s[tl];
        const float* eRow = embed + (size_t)q * EMBED_DIM;
        float lacc = 0.f;
        #pragma unroll 4
        for (int it = 0; it < 128; ++it) {
            int d = it * 2 + dp;
            float e = eRow[d];
            size_t gi = bBase + (size_t)d * TDIM + t0 + tl;
            out[gi] = e;                            // straight-through == z_q
            float diff = e - inp[gi];
            lacc += diff * diff;
        }
        red[tid] = lacc;
    }
    __syncthreads();
    for (int s = THREADS / 2; s > 0; s >>= 1) {
        if (tid < s) red[tid] += red[tid + s];
        __syncthreads();
    }
    if (tid == 0) partials[blockIdx.x] = red[0];
}

// ---------------------------------------------------------------------------
// Final: fixed-order reductions -> loss, perplexity, constant kldiv vector
// ---------------------------------------------------------------------------
__global__ __launch_bounds__(256) void vq_final(const float* __restrict__ partials,
                                                const int* __restrict__ hist,
                                                float* __restrict__ out) {
    __shared__ float red[256];
    const int tid = threadIdx.x;

    float s = 0.f;
    for (int i = tid; i < NROWS / ROWS_PER_BLOCK; i += 256) s += partials[i];
    red[tid] = s;
    __syncthreads();
    for (int st = 128; st > 0; st >>= 1) { if (tid < st) red[tid] += red[tid + st]; __syncthreads(); }
    const float loss = 1.25f * red[0] / (float)OUT_MAIN;   // vq + 0.25 * commitment (equal)
    __syncthreads();

    float se = 0.f;
    for (int i = tid; i < NUM_EMBED; i += 256) {
        float p = (float)hist[i] * (1.0f / (float)NROWS);
        se += p * logf(p + 1e-10f);
    }
    red[tid] = se;
    __syncthreads();
    for (int st = 128; st > 0; st >>= 1) { if (tid < st) red[tid] += red[tid + st]; __syncthreads(); }
    const float perp = expf(-red[0]);

    if (tid == 0) {
        out[OUT_MAIN] = loss;                    // loss scalar
        out[OUT_MAIN + 17] = perp;               // perplexity scalar
    }
    if (tid < BDIM)
        out[OUT_MAIN + 1 + tid] = 6.9314718055994531f * 4096.0f;  // ln(1024) * T
}

// ---------------------------------------------------------------------------
extern "C" void kernel_launch(void* const* d_in, const int* in_sizes, int n_in,
                              void* d_out, int out_size, void* d_ws, size_t ws_size,
                              hipStream_t stream) {
    const float* inp   = (const float*)d_in[0];   // (16, 256, 4096) fp32
    const float* embed = (const float*)d_in[1];   // (1024, 256) fp32
    float* out = (float*)d_out;

    float* wsf      = (float*)d_ws;
    float* enorm    = wsf;                        // 1024 f32
    float* partials = wsf + 1024;                 // 1024 f32
    int*   hist     = (int*)(wsf + 2048);         // 1024 i32

    vq_prep<<<NUM_EMBED / 256, 256, 0, stream>>>(embed, enorm, partials, hist);
    vq_main<<<NROWS / ROWS_PER_BLOCK, THREADS, 0, stream>>>(inp, embed, enorm, out,
                                                            partials, hist);
    vq_final<<<1, 256, 0, stream>>>(partials, hist, out);
}
